// SIGIR2_21199958573611
// MI455X (gfx1250) — matrix operand order repaired and checked by reading, not compiled, not run on hardware
//
#include <hip/hip_runtime.h>
#include <hip/hip_bf16.h>

// ---------------------------------------------------------------------------
// AutoCorrelation attention (Autoformer) for MI455X / gfx1250, wave32 + WMMA.
// B=8, L=4096, D=512, H=8, e=64, top_k=8.
// Correlation computed as diagonal sums of Q@K^T via bf16 WMMA tiles.
// Double-buffered LDS + async global->LDS staging (ASYNCcnt).
// ---------------------------------------------------------------------------

#define B_   8
#define L_   4096
#define D_   512
#define TOPK 8

typedef __attribute__((ext_vector_type(16))) __bf16 v16bf;
typedef __attribute__((ext_vector_type(8)))  float  v8f;

union FragAB { v16bf v; unsigned int u[8]; };

#define TS_M 128
#define TS_N 128
#define TS_K 32
#define SA   40   // LDS row stride (bf16 elems), 80 bytes (8B aligned chunks)
#define SB   40

#ifndef __has_builtin
#define __has_builtin(x) 0
#endif
#if __has_builtin(__builtin_amdgcn_global_load_async_to_lds_b64) && \
    __has_builtin(__builtin_amdgcn_s_wait_asynccnt)
#define HAVE_ASYNC 1
#else
#define HAVE_ASYNC 0
#endif

typedef int v2i_ __attribute__((vector_size(8)));   // 8-byte payload type

__device__ __forceinline__ unsigned short f2bf(float x) {
  unsigned int u = __float_as_uint(x);
  unsigned int r = u + 0x7FFFu + ((u >> 16) & 1u);   // round-to-nearest-even
  return (unsigned short)(r >> 16);
}
__device__ __forceinline__ float bf2f(unsigned int h) {
  return __uint_as_float((h & 0xFFFFu) << 16);
}

// Issue one async 8-byte global->LDS copy (ASYNCcnt-tracked).
__device__ __forceinline__ void async_copy_b64(const unsigned short* g,
                                               unsigned short* l) {
#if HAVE_ASYNC
  __builtin_amdgcn_global_load_async_to_lds_b64(
      (__attribute__((address_space(1))) v2i_*)(g),
      (__attribute__((address_space(3))) v2i_*)(l), 0, 0);
#else
  *(uint2*)l = *(const uint2*)g;
#endif
}

__device__ __forceinline__ void wait_async_8() {
#if HAVE_ASYNC
  __builtin_amdgcn_s_wait_asynccnt(8);
#endif
}
__device__ __forceinline__ void wait_async_4() {
#if HAVE_ASYNC
  __builtin_amdgcn_s_wait_asynccnt(4);
#endif
}
__device__ __forceinline__ void wait_async_0() {
#if HAVE_ASYNC
  __builtin_amdgcn_s_wait_asynccnt(0);
#endif
}

// Stage a 128-row x 32-elem bf16 tile into LDS (stride SA) from a row-major
// bf16 global slab. 4 async instructions per wave.
__device__ __forceinline__ void stage_bf16_tile(unsigned short* lds,
                                                const unsigned short* gsrc,
                                                int rowStride) {
  const int tid = threadIdx.x;
#pragma unroll
  for (int i = 0; i < 4; ++i) {
    int f   = tid + i * 256;       // 0..1023
    int row = f >> 3;              // 0..127
    int c   = (f & 7) << 2;        // elem offset 0,4,...,28 (8B chunks)
    async_copy_b64(gsrc + (size_t)row * rowStride + c, lds + row * SA + c);
  }
}

// Stage with fp32->bf16 conversion (synchronous; conversion needs VALU).
__device__ __forceinline__ void stage_f32_tile(unsigned short* lds,
                                               const float* gsrc,
                                               int rowStride) {
  const int tid = threadIdx.x;
#pragma unroll
  for (int i = 0; i < 4; ++i) {
    int f   = tid + i * 256;
    int row = f >> 3;
    int kq  = (f & 7) << 2;
    float4 d = *(const float4*)(gsrc + (size_t)row * rowStride + kq);
    ushort4 h;
    h.x = f2bf(d.x); h.y = f2bf(d.y); h.z = f2bf(d.z); h.w = f2bf(d.w);
    *(ushort4*)&lds[row * SA + kq] = h;
  }
}

// A fragment (16x32 bf16): lane holds row (lane&15); lanes<16 get K {0-7,16-23},
// lanes>=16 get K {8-15,24-31}; VGPR v holds K pair per ISA table.
__device__ __forceinline__ v16bf load_frag_A(const unsigned short* lds,
                                             int rowBase, int lane) {
  FragAB f;
  int row  = rowBase + (lane & 15);
  int half = lane >> 4;
  const unsigned short* p = lds + row * SA + half * 8;
#pragma unroll
  for (int v = 0; v < 8; ++v) {
    int koff = ((v & 3) << 1) + ((v >> 2) << 4);
    f.u[v] = *(const unsigned int*)(p + koff);
  }
  return f.v;
}

// B fragment (32x16 bf16) from N-major LDS tile Bt[n][k].
__device__ __forceinline__ v16bf load_frag_B(const unsigned short* lds,
                                             int colBase, int lane) {
  FragAB f;
  int col  = colBase + (lane & 15);
  int half = lane >> 4;
  const unsigned short* p = lds + col * SB + half * 8;
#pragma unroll
  for (int v = 0; v < 8; ++v) {
    f.u[v] = *(const unsigned int*)(p + ((v & 3) << 1) + ((v >> 2) << 4));
  }
  return f.v;
}

// ---------------------------------------------------------------------------
// Kernel: zero fp32 buffer
// ---------------------------------------------------------------------------
__global__ __launch_bounds__(256) void ac_zero_kernel(float* p, int n) {
  int i = blockIdx.x * 256 + threadIdx.x;
  if (i < n) p[i] = 0.0f;
}

// ---------------------------------------------------------------------------
// Kernel: convert + transpose one 512x512 fp32 weight -> bf16 Wt[n][k]
// ---------------------------------------------------------------------------
__global__ __launch_bounds__(256) void ac_wt_kernel(
    const float* __restrict__ W, unsigned short* __restrict__ Wt) {
  int e = blockIdx.x * 256 + threadIdx.x;   // 0 .. 512*512-1
  int k = e >> 9;
  int n = e & 511;
  Wt[(size_t)n * D_ + k] = f2bf(W[(size_t)k * D_ + n]);
}

// ---------------------------------------------------------------------------
// GEMM: C[MxN] = A[MxK] @ Wt^T + bias, with Wt pre-transposed bf16 [N][K].
// A fp32 (converted during staging) or bf16 (async-staged); C bf16 or fp32.
// Double-buffered LDS; async tile DMA overlapped with WMMA.
// ---------------------------------------------------------------------------
template <bool A_BF16, bool OUT_BF16>
__global__ __launch_bounds__(256) void ac_gemm_kernel(
    const void* __restrict__ Aptr, const unsigned short* __restrict__ Wt,
    const float* __restrict__ bias, void* __restrict__ Cptr,
    int M, int K, int N) {
  __shared__ unsigned short As[2][TS_M * SA];
  __shared__ unsigned short Bs[2][TS_N * SB];

  const int tid   = threadIdx.x;
  const int lane  = tid & 31;
  const int wave  = tid >> 5;
  const int waveM = wave & 1;   // 2 x 64-row halves
  const int waveN = wave >> 1;  // 4 x 32-col quarters
  const int mBase = blockIdx.y * TS_M;
  const int nBase = blockIdx.x * TS_N;

  const float*          Af = (const float*)Aptr;
  const unsigned short* Ab = (const unsigned short*)Aptr;

  v8f acc[4][2];
#pragma unroll
  for (int i = 0; i < 4; ++i)
#pragma unroll
    for (int j = 0; j < 2; ++j)
      acc[i][j] = (v8f){0.f, 0.f, 0.f, 0.f, 0.f, 0.f, 0.f, 0.f};

  // prologue: stage first K-slab into buffer 0
  if (A_BF16) stage_bf16_tile(As[0], Ab + (size_t)mBase * K, K);
  else        stage_f32_tile (As[0], Af + (size_t)mBase * K, K);
  stage_bf16_tile(Bs[0], Wt + (size_t)nBase * K, K);

  const int nsteps = K / TS_K;
  int buf = 0;
  for (int s = 0; s < nsteps; ++s) {
    const int  k0      = s * TS_K;
    const bool hasNext = (s + 1) < nsteps;
    if (hasNext) {
      if (A_BF16) stage_bf16_tile(As[buf ^ 1], Ab + (size_t)mBase * K + k0 + TS_K, K);
      else        stage_f32_tile (As[buf ^ 1], Af + (size_t)mBase * K + k0 + TS_K, K);
      stage_bf16_tile(Bs[buf ^ 1], Wt + (size_t)nBase * K + k0 + TS_K, K);
      if (A_BF16) wait_async_8();   // current tile's 8 copies complete
      else        wait_async_4();   // only B tile rides ASYNCcnt
    } else {
      wait_async_0();
    }
    __syncthreads();

    v16bf afr[4], bfr[2];
#pragma unroll
    for (int ti = 0; ti < 4; ++ti)
      afr[ti] = load_frag_A(As[buf], waveM * 64 + ti * 16, lane);
#pragma unroll
    for (int tj = 0; tj < 2; ++tj)
      bfr[tj] = load_frag_B(Bs[buf], waveN * 32 + tj * 16, lane);
#pragma unroll
    for (int ti = 0; ti < 4; ++ti)
#pragma unroll
      for (int tj = 0; tj < 2; ++tj)
        acc[ti][tj] = __builtin_amdgcn_wmma_f32_16x16x32_bf16(
            false, afr[ti], false, bfr[tj], (short)0, acc[ti][tj], false, false);
    __syncthreads();
    buf ^= 1;
  }

  // ---- epilogue: add bias, store ----
  const int half = lane >> 4;
  const int nl   = lane & 15;
  unsigned short* Cb = (unsigned short*)Cptr;
  float*          Cf = (float*)Cptr;
#pragma unroll
  for (int ti = 0; ti < 4; ++ti) {
#pragma unroll
    for (int tj = 0; tj < 2; ++tj) {
      int gn  = nBase + waveN * 32 + tj * 16 + nl;
      int gm0 = mBase + waveM * 64 + ti * 16 + half * 8;
      float bv = bias[gn];
#pragma unroll
      for (int i = 0; i < 8; ++i) {
        float val = acc[ti][tj][i] + bv;
        size_t off = (size_t)(gm0 + i) * N + gn;
        if (OUT_BF16) Cb[off] = f2bf(val);
        else          Cf[off] = val;
      }
    }
  }
}

// ---------------------------------------------------------------------------
// Correlation kernel: mean_value[b,tau] += (1/D) * sum over (i,j) with
// (i-j)%L == tau of (Q_b @ K_b^T)[i,j].  One 128x128 tile of G per block via
// bf16 WMMA (async double-buffered staging), collapsed onto 255 diagonals in
// LDS (ds_add_f32), then folded into global mean_value with atomics.
// grid = (L/128 [j], L/128 [i], B)
// ---------------------------------------------------------------------------
__global__ __launch_bounds__(256) void ac_corr_kernel(
    const unsigned short* __restrict__ Qb, const unsigned short* __restrict__ Kb,
    float* __restrict__ mean_value) {
  __shared__ unsigned short Qs[2][TS_M * SA];
  __shared__ unsigned short Ks[2][TS_N * SB];
  __shared__ float diag[256];

  const int tid   = threadIdx.x;
  const int lane  = tid & 31;
  const int wave  = tid >> 5;
  const int waveM = wave & 1;
  const int waveN = wave >> 1;
  const int jT    = blockIdx.x;
  const int iT    = blockIdx.y;
  const int b     = blockIdx.z;

  const unsigned short* Q  = Qb + (size_t)b * L_ * D_ + (size_t)(iT * TS_M) * D_;
  const unsigned short* Kp = Kb + (size_t)b * L_ * D_ + (size_t)(jT * TS_N) * D_;

  diag[tid] = 0.0f;

  v8f acc[4][2];
#pragma unroll
  for (int i = 0; i < 4; ++i)
#pragma unroll
    for (int j = 0; j < 2; ++j)
      acc[i][j] = (v8f){0.f, 0.f, 0.f, 0.f, 0.f, 0.f, 0.f, 0.f};

  stage_bf16_tile(Qs[0], Q, D_);
  stage_bf16_tile(Ks[0], Kp, D_);

  const int nsteps = D_ / TS_K;
  int buf = 0;
  for (int s = 0; s < nsteps; ++s) {
    const int  d0      = s * TS_K;
    const bool hasNext = (s + 1) < nsteps;
    if (hasNext) {
      stage_bf16_tile(Qs[buf ^ 1], Q + d0 + TS_K, D_);
      stage_bf16_tile(Ks[buf ^ 1], Kp + d0 + TS_K, D_);
      wait_async_8();
    } else {
      wait_async_0();
    }
    __syncthreads();

    v16bf afr[4], bfr[2];
#pragma unroll
    for (int ti = 0; ti < 4; ++ti)
      afr[ti] = load_frag_A(Qs[buf], waveM * 64 + ti * 16, lane);
#pragma unroll
    for (int tj = 0; tj < 2; ++tj)
      bfr[tj] = load_frag_B(Ks[buf], waveN * 32 + tj * 16, lane); // B[d][j]=k[j][d]
#pragma unroll
    for (int ti = 0; ti < 4; ++ti)
#pragma unroll
      for (int tj = 0; tj < 2; ++tj)
        acc[ti][tj] = __builtin_amdgcn_wmma_f32_16x16x32_bf16(
            false, afr[ti], false, bfr[tj], (short)0, acc[ti][tj], false, false);
    __syncthreads();
    buf ^= 1;
  }

  // ---- collapse tile onto diagonals: delta = local_m - local_n + 127 ----
  const int half = lane >> 4;
  const int nl   = lane & 15;
#pragma unroll
  for (int ti = 0; ti < 4; ++ti) {
#pragma unroll
    for (int tj = 0; tj < 2; ++tj) {
      int lm0 = waveM * 64 + ti * 16 + half * 8;
      int ln  = waveN * 32 + tj * 16 + nl;
#pragma unroll
      for (int i = 0; i < 8; ++i) {
        int delta = lm0 + i - ln + 127;          // 0..254
        atomicAdd(&diag[delta], acc[ti][tj][i]); // ds_add_f32
      }
    }
  }
  __syncthreads();

  if (tid < 255) {
    int tau = ((iT - jT) * TS_M + tid - 127) & (L_ - 1);
    atomicAdd(&mean_value[b * L_ + tau], diag[tid] * (1.0f / (float)D_));
  }
}

// ---------------------------------------------------------------------------
// Top-k (k=8) over batch-mean of mean_value, then per-batch softmax weights.
// ---------------------------------------------------------------------------
__global__ __launch_bounds__(256) void ac_topk_softmax_kernel(
    const float* __restrict__ mean_value, int* __restrict__ idx_out,
    float* __restrict__ w_out) {
  __shared__ float meanb[L_];
  __shared__ float rv[256];
  __shared__ int   ri[256];
  __shared__ int   sidx[TOPK];

  const int t = threadIdx.x;
  for (int i = t; i < L_; i += 256) {
    float s = 0.0f;
    for (int b = 0; b < B_; ++b) s += mean_value[b * L_ + i];
    meanb[i] = s;
  }
  __syncthreads();

  for (int r = 0; r < TOPK; ++r) {
    float best = -3.4e38f; int bi = 0;
    for (int i = t; i < L_; i += 256) {
      float v = meanb[i];
      if (v > best) { best = v; bi = i; }
    }
    rv[t] = best; ri[t] = bi;
    __syncthreads();
    if (t == 0) {
      float bb = rv[0]; int bj = ri[0];
      for (int i = 1; i < 256; ++i)
        if (rv[i] > bb) { bb = rv[i]; bj = ri[i]; }
      sidx[r] = bj;
      meanb[bj] = -3.4e38f;
    }
    __syncthreads();
  }

  if (t < TOPK) idx_out[t] = sidx[t];
  if (t < B_) {
    float w[TOPK]; float mx = -3.4e38f;
#pragma unroll
    for (int k = 0; k < TOPK; ++k) {
      w[k] = mean_value[t * L_ + sidx[k]];
      mx = fmaxf(mx, w[k]);
    }
    float s = 0.0f;
#pragma unroll
    for (int k = 0; k < TOPK; ++k) { w[k] = __expf(w[k] - mx); s += w[k]; }
    float inv = 1.0f / s;
#pragma unroll
    for (int k = 0; k < TOPK; ++k) w_out[t * TOPK + k] = w[k] * inv;
  }
}

// ---------------------------------------------------------------------------
// Aggregation: agg[b,l,d] = sum_k w[b,k] * v[b,(l+idx[k])%L, d]  (bf16 in/out)
// ---------------------------------------------------------------------------
__global__ __launch_bounds__(256) void ac_agg_kernel(
    const unsigned short* __restrict__ vbf, const int* __restrict__ idx,
    const float* __restrict__ w, unsigned short* __restrict__ aggbf) {
  const int p   = blockIdx.x * 256 + threadIdx.x;     // bf16-pair index
  const int LD2 = L_ * (D_ / 2);
  const int b   = p / LD2;
  const int rem = p % LD2;
  const int l   = rem >> 8;      // D_/2 = 256
  const int d2  = rem & 255;

  float s0 = 0.0f, s1 = 0.0f;
#pragma unroll
  for (int k = 0; k < TOPK; ++k) {
    int ls = l + idx[k];
    if (ls >= L_) ls -= L_;
    unsigned int dv =
        *(const unsigned int*)(vbf + ((size_t)b * L_ + ls) * D_ + d2 * 2);
    float wv = w[b * TOPK + k];
    s0 += wv * bf2f(dv);
    s1 += wv * bf2f(dv >> 16);
  }
  unsigned int out = (unsigned int)f2bf(s0) | ((unsigned int)f2bf(s1) << 16);
  *(unsigned int*)(aggbf + (size_t)p * 2) = out;
}

// ---------------------------------------------------------------------------
// Host-side orchestration
// ---------------------------------------------------------------------------
extern "C" void kernel_launch(void* const* d_in, const int* in_sizes, int n_in,
                              void* d_out, int out_size, void* d_ws, size_t ws_size,
                              hipStream_t stream) {
  (void)in_sizes; (void)n_in; (void)out_size; (void)ws_size;
  const float* queries = (const float*)d_in[0];
  const float* keys    = (const float*)d_in[1];
  const float* values  = (const float*)d_in[2];
  const float* Wq = (const float*)d_in[3];
  const float* bq = (const float*)d_in[4];
  const float* Wk = (const float*)d_in[5];
  const float* bk = (const float*)d_in[6];
  const float* Wv = (const float*)d_in[7];
  const float* bv = (const float*)d_in[8];
  const float* Wo = (const float*)d_in[9];
  const float* bo = (const float*)d_in[10];

  const size_t S  = (size_t)B_ * L_ * D_ * sizeof(unsigned short); // 33.5 MB
  const size_t WM = (size_t)D_ * D_;                               // elems/weight
  char* ws = (char*)d_ws;
  unsigned short* qbf   = (unsigned short*)(ws);
  unsigned short* kbf   = (unsigned short*)(ws + S);
  unsigned short* vbf   = (unsigned short*)(ws + 2 * S);
  unsigned short* aggbf = (unsigned short*)(ws + 3 * S);
  float*          meanv = (float*)(ws + 4 * S);
  int*            idxp  = (int*)(ws + 4 * S + (size_t)B_ * L_ * sizeof(float));
  float*          wgt   = (float*)((char*)idxp + 64);
  unsigned short* WtQ   = (unsigned short*)(ws + 4 * S + (1 << 18));
  unsigned short* WtK   = WtQ + WM;
  unsigned short* WtV   = WtQ + 2 * WM;
  unsigned short* WtO   = WtQ + 3 * WM;

  const int M = B_ * L_;   // 32768

  // 1) zero correlation accumulator; convert+transpose weights to bf16 Wt[n][k]
  ac_zero_kernel<<<(B_ * L_ + 255) / 256, 256, 0, stream>>>(meanv, B_ * L_);
  ac_wt_kernel<<<(D_ * D_) / 256, 256, 0, stream>>>(Wq, WtQ);
  ac_wt_kernel<<<(D_ * D_) / 256, 256, 0, stream>>>(Wk, WtK);
  ac_wt_kernel<<<(D_ * D_) / 256, 256, 0, stream>>>(Wv, WtV);
  ac_wt_kernel<<<(D_ * D_) / 256, 256, 0, stream>>>(Wo, WtO);

  // 2) Q/K/V projections (fp32 in, bf16 out) — WMMA GEMMs
  dim3 gGemm(D_ / TS_N, M / TS_M);  // (4, 256)
  ac_gemm_kernel<false, true><<<gGemm, 256, 0, stream>>>(queries, WtQ, bq, qbf, M, D_, D_);
  ac_gemm_kernel<false, true><<<gGemm, 256, 0, stream>>>(keys,    WtK, bk, kbf, M, D_, D_);
  ac_gemm_kernel<false, true><<<gGemm, 256, 0, stream>>>(values,  WtV, bv, vbf, M, D_, D_);

  // 3) autocorrelation scores via diagonal-summed Q@K^T (WMMA)
  dim3 gCorr(L_ / TS_N, L_ / TS_M, B_);  // (32, 32, 8)
  ac_corr_kernel<<<gCorr, 256, 0, stream>>>(qbf, kbf, meanv);

  // 4) top-8 lags + per-batch softmax weights
  ac_topk_softmax_kernel<<<1, 256, 0, stream>>>(meanv, idxp, wgt);

  // 5) shift-gather weighted aggregation of V
  ac_agg_kernel<<<(B_ * L_ * (D_ / 2)) / 256, 256, 0, stream>>>(vbf, idxp, wgt, aggbf);

  // 6) output projection (bf16 in, fp32 out) — WMMA GEMM
  ac_gemm_kernel<true, false><<<gGemm, 256, 0, stream>>>(aggbf, WtO, bo, d_out, M, D_, D_);
}
